// Attn_128849019039
// MI455X (gfx1250) — matrix operand order repaired and checked by reading, not compiled, not run on hardware
//
#include <hip/hip_runtime.h>

// ---------------------------------------------------------------------------
// Causal MHA forward for MI455X (gfx1250), wave32 + v_wmma_f32_16x16x32_f16.
// B=4, T=4096, D_MODEL=128, N_HEADS=4, HEAD_DIM=32.
// Attention is computed in TRANSPOSED orientation (S^T, O^T) so softmax
// reductions are in-register (rows = keys = VGPRs, cols = queries = lanes).
// ---------------------------------------------------------------------------

typedef _Float16 f16;
typedef __attribute__((ext_vector_type(16))) _Float16 v16h;
typedef __attribute__((ext_vector_type(8)))  _Float16 v8h;
typedef __attribute__((ext_vector_type(8)))  float    v8f;
typedef __attribute__((ext_vector_type(4)))  float    v4f;

#define DM      128
#define NH      4
#define HD      32
#define BATCH   4
#define TSEQ    4096
#define BT      (BATCH * TSEQ)          // 16384 tokens
#define QK_SCALE 0.17677669529663687f   // 1/sqrt(32)

union V16H { v16h v; v8h h[2]; f16 e[16]; };
union V8H  { v8h  v; f16 e[8]; };

__device__ __forceinline__ v8f wmma_f16(v16h a, v16h b, v8f c) {
  // D = A(16x32 f16) * B(32x16 f16) + C(16x16 f32)
  return __builtin_amdgcn_wmma_f32_16x16x32_f16(
      /*neg_a=*/false, a, /*neg_b=*/false, b,
      /*c_mod=*/(short)0, c, /*reuse_a=*/false, /*reuse_b=*/false);
}

// ---------------------------------------------------------------------------
// Kernel 1: qkv = x @ W_qkv^T  (M=16384, N=384, K=128), f32 in -> f16 out.
// Writes Q (pre-scaled) and K row-major [bh][t][32]; V transposed [bh][32][t].
// One wave per 16x16 output tile; 4 WMMAs (K=128).
// ---------------------------------------------------------------------------
__global__ __launch_bounds__(256) void qkv_proj_kernel(
    const float* __restrict__ x, const float* __restrict__ Wqkv,
    f16* __restrict__ Qh, f16* __restrict__ Kh, f16* __restrict__ Vt) {
  const int lane = threadIdx.x & 31;
  const int wave = threadIdx.x >> 5;
  const int w    = blockIdx.x * 8 + wave;       // 24576 waves total
  const int mtile = w / 24, ntile = w % 24;
  const int mbase = mtile * 16, nbase = ntile * 16;
  const int lm = lane & 15, hf = lane >> 4;

  const float* arow = x    + (size_t)(mbase + lm) * DM; // A: x row (M = lm)
  const float* brow = Wqkv + (size_t)(nbase + lm) * DM; // B: weight row (N = lm)

  v8f c = {};
  for (int kk = 0; kk < DM; kk += 32) {
    V16H a, b;
    // A layout: 16 consecutive K values starting at kk + 16*hf
    const v4f* pa0 = (const v4f*)(arow + kk + 16 * hf);
    const v4f* pa1 = (const v4f*)(arow + kk + 16 * hf + 8);
    v4f a0 = pa0[0], a1 = pa0[1], a2 = pa1[0], a3 = pa1[1];
    // B layout: 16 consecutive K values starting at kk + 16*hf
    const v4f* pb = (const v4f*)(brow + kk + 16 * hf);
    v4f b0 = pb[0], b1 = pb[1], b2 = pb[2], b3 = pb[3];
#pragma unroll
    for (int i = 0; i < 4; ++i) {
      a.e[i]      = (f16)a0[i]; a.e[4 + i]  = (f16)a1[i];
      a.e[8 + i]  = (f16)a2[i]; a.e[12 + i] = (f16)a3[i];
      b.e[i]      = (f16)b0[i]; b.e[4 + i]  = (f16)b1[i];
      b.e[8 + i]  = (f16)b2[i]; b.e[12 + i] = (f16)b3[i];
    }
    c = wmma_f16(a.v, b.v, c);
  }

  // C layout: col = lm, row = r + 8*hf. 'which'/'hh' are tile-uniform.
  const int gn = nbase + lm;
  const int which = gn >> 7, hh = (gn >> 5) & 3, d = gn & 31;
  const int g0 = mbase + 8 * hf;                // first of 8 consecutive tokens
  const int bb = g0 >> 12, t0 = g0 & (TSEQ - 1);
  const int bh = bb * NH + hh;
  if (which == 0) {
#pragma unroll
    for (int r = 0; r < 8; ++r)
      Qh[(size_t)(bh * TSEQ + t0 + r) * HD + d] = (f16)(c[r] * QK_SCALE);
  } else if (which == 1) {
#pragma unroll
    for (int r = 0; r < 8; ++r)
      Kh[(size_t)(bh * TSEQ + t0 + r) * HD + d] = (f16)c[r];
  } else {
    V8H pk;                                     // 8 consecutive t: one b128
#pragma unroll
    for (int r = 0; r < 8; ++r) pk.e[r] = (f16)c[r];
    *(v8h*)(Vt + (size_t)(bh * HD + d) * TSEQ + t0) = pk.v;
  }
}

// ---------------------------------------------------------------------------
// Kernel 2: flash attention, transposed. One wave = one (b, h, 16-query tile).
// Per 32-key tile:  S^T = K_tile x Q^T (2 WMMAs) -> per-lane softmax
// (in-register reduction + one shfl_xor(16)) -> P^T via 2 ds_store_b128 /
// 2 ds_load_b128 -> O^T += V^T x P^T (2 WMMAs).
// ---------------------------------------------------------------------------
__global__ __launch_bounds__(256) void attn_kernel(
    const f16* __restrict__ Qh, const f16* __restrict__ Kh,
    const f16* __restrict__ Vt, f16* __restrict__ Oa) {
  __shared__ __align__(16) f16 lds[8][16 * 32];   // 1 KB per wave
  const int lane = threadIdx.x & 31;
  const int wave = threadIdx.x >> 5;
  const int w     = blockIdx.x * 8 + wave;        // 4096 waves total
  const int qtile = w & 255;                      // T/16 = 256
  const int bh    = w >> 8;                       // B*H = 16
  const int qbase = qtile * 16;
  const int lm = lane & 15, hf = lane >> 4;
  f16* P = lds[wave];                             // P^T as [q][k] (16x32)

  const f16* Qb = Qh + (size_t)bh * TSEQ * HD;
  const f16* Kb = Kh + (size_t)bh * TSEQ * HD;
  const f16* Vb = Vt + (size_t)bh * HD * TSEQ;

  // Q^T B-operand (held): lane = query column, dims 8*hf.. and 16+8*hf..
  V16H qt;
  { const f16* p = Qb + (size_t)(qbase + lm) * HD;
    qt.h[0] = *(const v8h*)(p + 8 * hf); qt.h[1] = *(const v8h*)(p + 16 + 8 * hf); }

  // Per-lane softmax state for query column q = qbase + lm.
  float m_i = -1e30f, l_i = 0.0f;
  v8f o0 = {}, o1 = {};   // O^T: rows d = r+8*hf (o0) / 16+r+8*hf (o1), col q

  const int ntiles = (qbase + 47) >> 5;           // causal: keys <= qbase+15
  for (int kt = 0; kt < ntiles; ++kt) {
    const int kbase = kt * 32;

    if (kt + 1 < ntiles) {                        // prefetch next K/V tiles
      __builtin_prefetch(Kb + (size_t)(kbase + 32 + lm) * HD, 0, 1);
      __builtin_prefetch(Vb + (size_t)lm * TSEQ + kbase + 32, 0, 1);
    }

    // K A-operands: rows = keys, dims contiguous.
    V16H ka0, ka1;
    ka0.h[0] = *(const v8h*)(Kb + (size_t)(kbase + lm) * HD + 8 * hf);
    ka0.h[1] = *(const v8h*)(Kb + (size_t)(kbase + lm) * HD + 16 + 8 * hf);
    ka1.h[0] = *(const v8h*)(Kb + (size_t)(kbase + 16 + lm) * HD + 8 * hf);
    ka1.h[1] = *(const v8h*)(Kb + (size_t)(kbase + 16 + lm) * HD + 16 + 8 * hf);

    v8f st0 = {}, st1 = {};
    st0 = wmma_f16(ka0.v, qt.v, st0);   // keys kbase    + (r + 8*hf)
    st1 = wmma_f16(ka1.v, qt.v, st1);   // keys kbase+16 + (r + 8*hf)

    if (kbase + 31 > qbase) {           // diagonal tile: mask k > q
      const int q = qbase + lm;
#pragma unroll
      for (int r = 0; r < 8; ++r) {
        if (kbase + r + 8 * hf > q)      st0[r] = -1e30f;
        if (kbase + 16 + r + 8 * hf > q) st1[r] = -1e30f;
      }
    }

    // Column (=lane) softmax: max/sum over 16 regs, merge halves via xor 16.
    float mx = fmaxf(st0[0], st1[0]);
#pragma unroll
    for (int r = 1; r < 8; ++r) mx = fmaxf(mx, fmaxf(st0[r], st1[r]));
    mx = fmaxf(mx, __shfl_xor(mx, 16, 32));
    const float mnew  = fmaxf(m_i, mx);
    const float alpha = __expf(m_i - mnew);

    V8H p0, p1;
    float ps = 0.0f;
#pragma unroll
    for (int r = 0; r < 8; ++r) {
      const float e0 = __expf(st0[r] - mnew);
      const float e1 = __expf(st1[r] - mnew);
      ps += e0 + e1;
      p0.e[r] = (f16)e0;                // k_local = r + 8*hf
      p1.e[r] = (f16)e1;                // k_local = 16 + r + 8*hf
    }
    ps += __shfl_xor(ps, 16, 32);
    l_i = l_i * alpha + ps;
    m_i = mnew;
#pragma unroll
    for (int r = 0; r < 8; ++r) { o0[r] *= alpha; o1[r] *= alpha; }

    // P^T -> LDS [q][k]: per-lane values are contiguous in k => b128 stores.
    *(v8h*)(P + lm * 32 + 8 * hf)      = p0.v;
    *(v8h*)(P + lm * 32 + 16 + 8 * hf) = p1.v;
    __builtin_amdgcn_wave_barrier();              // LDS in-order within wave
    V16H pt;                                      // B-operand: K=key, N=q
    { pt.h[0] = *(const v8h*)(P + lm * 32 + 8 * hf);
      pt.h[1] = *(const v8h*)(P + lm * 32 + 16 + 8 * hf); }
    __builtin_amdgcn_wave_barrier();

    // V^T A-operands: rows = dims, keys contiguous (layout [bh][d][t]).
    V16H va0, va1;
    va0.h[0] = *(const v8h*)(Vb + (size_t)lm * TSEQ + kbase + 8 * hf);
    va0.h[1] = *(const v8h*)(Vb + (size_t)lm * TSEQ + kbase + 16 + 8 * hf);
    va1.h[0] = *(const v8h*)(Vb + (size_t)(16 + lm) * TSEQ + kbase + 8 * hf);
    va1.h[1] = *(const v8h*)(Vb + (size_t)(16 + lm) * TSEQ + kbase + 16 + 8 * hf);

    o0 = wmma_f16(va0.v, pt.v, o0);     // dims 0..15
    o1 = wmma_f16(va1.v, pt.v, o1);     // dims 16..31
  }

  // Normalize, pack, store: per lane one query column, 2x b128 stores.
  const float inv = 1.0f / l_i;
  V8H r0, r1;
#pragma unroll
  for (int r = 0; r < 8; ++r) {
    r0.e[r] = (f16)(o0[r] * inv);       // d = r + 8*hf
    r1.e[r] = (f16)(o1[r] * inv);       // d = 16 + r + 8*hf
  }
  const int bb = bh >> 2, hh = bh & 3;
  const int t = qbase + lm;
  const size_t base = ((size_t)(bb * TSEQ + t) * NH + hh) * HD;
  *(v8h*)(Oa + base + 8 * hf)      = r0.v;
  *(v8h*)(Oa + base + 16 + 8 * hf) = r1.v;
}

// ---------------------------------------------------------------------------
// Kernel 3: out = Oa @ W_out^T  (M=16384, N=128, K=128), f16 A, f32 result.
// ---------------------------------------------------------------------------
__global__ __launch_bounds__(256) void out_proj_kernel(
    const f16* __restrict__ Oa, const float* __restrict__ Wout,
    float* __restrict__ out) {
  const int lane = threadIdx.x & 31;
  const int wave = threadIdx.x >> 5;
  const int w = blockIdx.x * 8 + wave;            // 8192 waves total
  const int mtile = w >> 3, ntile = w & 7;
  const int mbase = mtile * 16, nbase = ntile * 16;
  const int lm = lane & 15, hf = lane >> 4;

  const f16*   arow = Oa   + (size_t)(mbase + lm) * DM;
  const float* brow = Wout + (size_t)(nbase + lm) * DM;

  v8f c = {};
  for (int kk = 0; kk < DM; kk += 32) {
    V16H a, b;
    a.h[0] = *(const v8h*)(arow + kk + 16 * hf);
    a.h[1] = *(const v8h*)(arow + kk + 16 * hf + 8);
    const v4f* pb = (const v4f*)(brow + kk + 16 * hf);
    v4f b0 = pb[0], b1 = pb[1], b2 = pb[2], b3 = pb[3];
#pragma unroll
    for (int i = 0; i < 4; ++i) {
      b.e[i]      = (f16)b0[i]; b.e[4 + i]  = (f16)b1[i];
      b.e[8 + i]  = (f16)b2[i]; b.e[12 + i] = (f16)b3[i];
    }
    c = wmma_f16(a.v, b.v, c);
  }
#pragma unroll
  for (int r = 0; r < 8; ++r) {
    const int g = mbase + r + 8 * hf;
    out[(size_t)g * DM + nbase + lm] = c[r];
  }
}

// ---------------------------------------------------------------------------
extern "C" void kernel_launch(void* const* d_in, const int* in_sizes, int n_in,
                              void* d_out, int out_size, void* d_ws, size_t ws_size,
                              hipStream_t stream) {
  const float* x    = (const float*)d_in[0];   // (4, 4096, 128)
  const float* Wqkv = (const float*)d_in[1];   // (384, 128)
  const float* Wout = (const float*)d_in[2];   // (128, 128)
  float* out = (float*)d_out;                  // (4, 4096, 128)

  // Workspace: 4 x 4 MiB f16 buffers = 16 MiB.
  f16* ws = (f16*)d_ws;
  const size_t SEG = (size_t)BATCH * NH * TSEQ * HD;   // 2,097,152 halves
  f16* Qh = ws;
  f16* Kh = ws + SEG;
  f16* Vt = ws + 2 * SEG;
  f16* Oa = ws + 3 * SEG;

  // 24576 waves: (16384/16) * (384/16) tiles, 8 waves per 256-thread block.
  qkv_proj_kernel<<<3072, 256, 0, stream>>>(x, Wqkv, Qh, Kh, Vt);
  // 4096 waves: B*H * (T/16) query tiles.
  attn_kernel<<<512, 256, 0, stream>>>(Qh, Kh, Vt, Oa);
  // 8192 waves: (16384/16) * (128/16) tiles.
  out_proj_kernel<<<1024, 256, 0, stream>>>(Oa, Wout, out);
}
